// sLSTMCell_37374805409861
// MI455X (gfx1250) — matrix operand order repaired and checked by reading, not compiled, not run on hardware
//
#include <hip/hip_runtime.h>
#include <hip/hip_bf16.h>

// sLSTM: B=8, T=2048, D=1024, H=1024.
// Phase 1: pre = x @ W_comb (+b)  -> 16384x4096 f32 in workspace (bf16 WMMA GEMM).
// Phase 2: persistent scan kernel, W_rec slices resident in LDS (320KB/WGP),
//          32x v_wmma_f32_16x16x32_bf16 per wave per step, h staged to LDS via
//          GLOBAL_LOAD_ASYNC_TO_LDS_B128 (ASYNCcnt engine), grid barrier per step.

#define Bsz 8
#define Tlen 2048
#define Ddim 1024
#define Hdim 1024
#define G4H 4096

typedef __attribute__((ext_vector_type(16))) __bf16 bf16x16;
typedef __attribute__((ext_vector_type(8)))  float  f32x8;
typedef __attribute__((ext_vector_type(4)))  int    v4i;

#if __has_builtin(__builtin_amdgcn_global_load_async_to_lds_b128) && \
    __has_builtin(__builtin_amdgcn_s_wait_asynccnt)
#define HAVE_ASYNC_LDS 1
typedef __attribute__((address_space(1))) v4i v4i_g;   // global v4i
typedef __attribute__((address_space(3))) v4i v4i_l;   // LDS v4i
#endif

union BfFrag {
    bf16x16 v;
    uint4   q[2];   // q[0] = VGPR0-3 (first K-group of 8), q[1] = VGPR4-7 (second group)
};

union BfPack2 {
    __bf16 h[2];
    unsigned u;
};

// ---------------------------------------------------------------------------
// Phase 1: pre[r][g] = sum_k x[r][k] * W_comb[k][g] + b_comb[g], r = b*T + t
// Block: 256 threads (8 waves). Tile: 16(M) x 128(N), K staged 32 at a time.
// ---------------------------------------------------------------------------
__global__ void gemm_pre_kernel(const float* __restrict__ x,
                                const float* __restrict__ Wc,
                                const float* __restrict__ bc,
                                float* __restrict__ pre) {
    __shared__ __bf16 lA[16 * 32];    // A tile, row-major [m][k]
    __shared__ __bf16 lB[128 * 32];   // B tile, col-major [n][k]

    const int tid  = threadIdx.x;
    const int wave = tid >> 5;        // 0..7 -> N sub-tile
    const int lane = tid & 31;
    const int ln   = lane & 15;
    const int hi   = lane >> 4;       // 0 or 1
    const int m0   = blockIdx.y * 16;
    const int n0   = blockIdx.x * 128;

    f32x8 acc = {};
    for (int kt = 0; kt < Ddim; kt += 32) {
        // Stage A: 16x32 f32 -> bf16; one float2 load + packed 4B store per thread
        {
            int i2 = tid * 2;                 // 0..510, even
            int m = i2 >> 5, k = i2 & 31;
            float2 f = *(const float2*)&x[(size_t)(m0 + m) * Ddim + kt + k];
            BfPack2 p; p.h[0] = (__bf16)f.x; p.h[1] = (__bf16)f.y;
            *(unsigned*)&lA[i2] = p.u;
        }
        // Stage B: 32x128 f32 -> bf16, transposed to [n][k]; float4 global loads
        for (int j = 0; j < 4; ++j) {
            int idx = tid * 16 + j * 4;       // 4096 elements total
            int k = idx >> 7, n = idx & 127;
            float4 w = *(const float4*)&Wc[(size_t)(kt + k) * G4H + n0 + n];
            lB[(n + 0) * 32 + k] = (__bf16)w.x;
            lB[(n + 1) * 32 + k] = (__bf16)w.y;
            lB[(n + 2) * 32 + k] = (__bf16)w.z;
            lB[(n + 3) * 32 + k] = (__bf16)w.w;
        }
        __syncthreads();

        // A fragment: lane ln = row; 16 contiguous K at hi*16
        BfFrag a, b;
        const uint4* rowA = (const uint4*)&lA[ln * 32];
        a.q[0] = rowA[2 * hi];
        a.q[1] = rowA[2 * hi + 1];
        // B fragment: col = wave*16+ln, 16 contiguous K at hi*16
        const uint4* colB = (const uint4*)&lB[(wave * 16 + ln) * 32 + hi * 16];
        b.q[0] = colB[0];
        b.q[1] = colB[1];

        acc = __builtin_amdgcn_wmma_f32_16x16x32_bf16(false, a.v, false, b.v,
                                                      (short)0, acc, false, false);
        __syncthreads();
    }

    // C layout: acc[v] -> row v + 8*hi, col ln
    const int col = n0 + wave * 16 + ln;
    const float bias = bc[col];
    for (int v = 0; v < 8; ++v) {
        int m = v + 8 * hi;
        pre[(size_t)(m0 + m) * G4H + col] = acc[v] + bias;
    }
}

// ---------------------------------------------------------------------------
// Grid barrier (sense-reversal via generation counter)
// ---------------------------------------------------------------------------
__device__ __forceinline__ void grid_barrier(unsigned* counter, unsigned* gen,
                                             unsigned nwg) {
    __threadfence();
    __syncthreads();
    if (threadIdx.x == 0) {
        unsigned g = __atomic_load_n(gen, __ATOMIC_ACQUIRE);
        unsigned arrived = atomicAdd(counter, 1u) + 1u;
        if (arrived == nwg) {
            __atomic_store_n(counter, 0u, __ATOMIC_RELAXED);
            __threadfence();
            __atomic_store_n(gen, g + 1u, __ATOMIC_RELEASE);
        } else {
            while (__atomic_load_n(gen, __ATOMIC_ACQUIRE) == g) {
                __builtin_amdgcn_s_sleep(1);
            }
        }
    }
    __syncthreads();
}

// ---------------------------------------------------------------------------
// Phase 2: persistent scan. 64 WGs x 128 threads (4 waves = 4 gates).
// WG owns hidden cols [j0, j0+16). LDS: W_rec slice (bf16 col-major, 128KB),
// h stage buffer (16KB, filled by async-to-LDS engine), gate tiles, c/n state.
// ---------------------------------------------------------------------------
__global__ void slstm_scan_kernel(const float* __restrict__ pre,
                                  const float* __restrict__ Wr,
                                  float* __restrict__ out,
                                  __bf16* __restrict__ h0buf,
                                  __bf16* __restrict__ h1buf,
                                  unsigned* __restrict__ bar) {
    extern __shared__ char smem[];
    __bf16 (*Wsl)[16][1024] = (__bf16 (*)[16][1024])smem;          // [4][16][1024] 128KB
    __bf16* hlds = (__bf16*)(smem + 4 * 16 * 1024 * sizeof(__bf16)); // [8][1024] 16KB
    float* gate = (float*)(hlds + Bsz * Hdim);                       // [4][8][16]
    float* cst  = gate + 4 * 8 * 16;                                 // [8][16]
    float* nst  = cst + 8 * 16;                                      // [8][16]

    const int tid  = threadIdx.x;   // 0..127
    const int wave = tid >> 5;      // gate id 0..3
    const int lane = tid & 31;
    const int ln   = lane & 15;
    const int hi   = lane >> 4;
    const int j0   = blockIdx.x * 16;
    const unsigned nwg = gridDim.x;

    // One-time: W_rec slice -> LDS, bf16, layout Wsl[g][n][k]
    for (int i = tid; i < 4 * 16 * 1024; i += 128) {
        int g = i >> 14;
        int n = (i >> 10) & 15;
        int k = i & 1023;
        Wsl[g][n][k] = (__bf16)Wr[(size_t)k * G4H + g * Hdim + j0 + n];
    }
    {   // init c = 0, n = 1
        int b = tid >> 4, n = tid & 15;
        cst[b * 16 + n] = 0.0f;
        nst[b * 16 + n] = 1.0f;
    }
    __syncthreads();

    const __bf16* hread  = h0buf;   // zero-initialized by init kernel
    __bf16*       hwrite = h1buf;

    const int mrow = ln & 7;        // A rows 8..15 duplicate batch rows (C rows 8..15 ignored)

    for (int t = 0; t < Tlen; ++t) {
        // ---- stage h(t) (8x1024 bf16 = 16KB) global -> LDS ----
#ifdef HAVE_ASYNC_LDS
        for (int r = 0; r < 8; ++r) {
            __builtin_amdgcn_global_load_async_to_lds_b128(
                (v4i_g*)(hread + r * 1024 + tid * 8),
                (v4i_l*)(hlds + r * 1024 + tid * 8), 0, 0);
        }
        __builtin_amdgcn_s_wait_asynccnt(0);
#else
        for (int r = 0; r < 8; ++r) {
            int i = r * 1024 + tid * 8;       // 8 bf16 = 16 bytes per copy
            *(uint4*)&hlds[i] = *(const uint4*)&hread[i];
        }
#endif
        __syncthreads();

        // ---- gates_pre_tile = h @ W_rec[:, gate*H + j0 : +16] ----
        f32x8 acc = {};
        for (int kt = 0; kt < Hdim; kt += 32) {
            BfFrag a, b;
            const uint4* rowH = (const uint4*)&hlds[mrow * Hdim + kt];
            a.q[0] = rowH[2 * hi];      // K kt+hi*16+{0..7}
            a.q[1] = rowH[2 * hi + 1];  // K kt+hi*16+{8..15}
            const uint4* colW = (const uint4*)&Wsl[wave][ln][kt + hi * 16];
            b.q[0] = colW[0];
            b.q[1] = colW[1];
            acc = __builtin_amdgcn_wmma_f32_16x16x32_bf16(false, a.v, false, b.v,
                                                          (short)0, acc, false, false);
        }

        // ---- add pre(t) and publish gate tiles to LDS ----
        if (lane < 16) {
            for (int v = 0; v < 8; ++v) {   // v = batch row
                float p = pre[((size_t)v * Tlen + t) * G4H + wave * Hdim + j0 + ln];
                gate[(wave * 8 + v) * 16 + ln] = acc[v] + p;
            }
            if (t + 1 < Tlen) {
                __builtin_prefetch(&pre[((size_t)(ln & 7) * Tlen + (t + 1)) * G4H +
                                        wave * Hdim + j0], 0, 1);
            }
        }
        __syncthreads();

        // ---- pointwise sLSTM update: thread -> (b, n) ----
        {
            int b = tid >> 4, n = tid & 15;
            float ig = __expf(gate[(0 * 8 + b) * 16 + n]);
            float fg = __expf(gate[(1 * 8 + b) * 16 + n]);
            float og = 1.0f / (1.0f + __expf(-gate[(2 * 8 + b) * 16 + n]));
            float ct = tanhf(gate[(3 * 8 + b) * 16 + n]);
            float nv = fg * nst[b * 16 + n] + ig;
            float cv = fg * cst[b * 16 + n] + ig * ct;
            nst[b * 16 + n] = nv;
            cst[b * 16 + n] = cv;
            float hv = og * (cv / fmaxf(nv, 1.0f));
            out[((size_t)b * Tlen + t) * Hdim + j0 + n] = hv;
            hwrite[b * Hdim + j0 + n] = (__bf16)hv;
            if (t == Tlen - 1) {
                size_t base = (size_t)Bsz * Tlen * Hdim;
                out[base + (size_t)b * Hdim + j0 + n] = hv;
                out[base + (size_t)Bsz * Hdim + (size_t)b * Hdim + j0 + n] = cv;
                out[base + 2 * (size_t)Bsz * Hdim + (size_t)b * Hdim + j0 + n] = nv;
            }
        }

        // ---- publish h(t+1), swap double buffers ----
        grid_barrier(bar, bar + 1, nwg);
        const __bf16* tmp = hread;
        hread  = hwrite;
        hwrite = (__bf16*)tmp;
    }
}

// ---------------------------------------------------------------------------
// Init: zero h double-buffers and barrier words (runs every launch -> graph-safe)
// ---------------------------------------------------------------------------
__global__ void init_kernel(__bf16* h0, __bf16* h1, unsigned* bar) {
    int i = blockIdx.x * blockDim.x + threadIdx.x;
    if (i < Bsz * Hdim) {
        h0[i] = (__bf16)0.0f;
        h1[i] = (__bf16)0.0f;
    }
    if (i < 2) bar[i] = 0u;
}

extern "C" void kernel_launch(void* const* d_in, const int* in_sizes, int n_in,
                              void* d_out, int out_size, void* d_ws, size_t ws_size,
                              hipStream_t stream) {
    (void)in_sizes; (void)n_in; (void)out_size; (void)ws_size;
    const float* x  = (const float*)d_in[0];   // (B,T,D)
    const float* Wc = (const float*)d_in[1];   // (D,4H)
    const float* bc = (const float*)d_in[2];   // (4H)
    const float* Wr = (const float*)d_in[3];   // (H,4H)
    float* out = (float*)d_out;                // (B,T,H) ++ h ++ c ++ n

    char* ws = (char*)d_ws;
    float* pre = (float*)ws;                                   // 16384*4096 f32 = 256MB
    size_t preBytes = (size_t)Bsz * Tlen * G4H * sizeof(float);
    __bf16* h0 = (__bf16*)(ws + preBytes);
    __bf16* h1 = h0 + Bsz * Hdim;
    unsigned* bar = (unsigned*)(ws + preBytes + 65536);

    init_kernel<<<32, 256, 0, stream>>>(h0, h1, bar);

    dim3 g1(G4H / 128, (Bsz * Tlen) / 16);   // (32, 1024)
    gemm_pre_kernel<<<g1, 256, 0, stream>>>(x, Wc, bc, pre);

    size_t smem2 = (size_t)4 * 16 * 1024 * sizeof(__bf16)      // 128KB W_rec slice
                 + (size_t)Bsz * Hdim * sizeof(__bf16)         // 16KB h stage
                 + (4 * 8 * 16 + 2 * 8 * 16) * sizeof(float);  // gates + c,n state
    slstm_scan_kernel<<<Hdim / 16, 128, smem2, stream>>>(pre, Wr, out, h0, h1, bar);
}
